// BaseModel_44341242364529
// MI455X (gfx1250) — matrix operand order repaired and checked by reading, not compiled, not run on hardware
//
#include <hip/hip_runtime.h>
#include <hip/hip_bf16.h>
#include <math.h>

typedef __attribute__((ext_vector_type(16))) __bf16 v16bf;
typedef __attribute__((ext_vector_type(8)))  __bf16 v8bf;
typedef __attribute__((ext_vector_type(8)))  float  v8f;

#define IMGX   64
#define GTD    16
#define NPATCH 4096
#define DMODEL 256
#define NHEAD  4
#define DHEAD  64
#define LSEQ   4098
#define DFFN   1024
#define PVOL   64
#define NLAY   2

// ---------- helpers ----------
__device__ __forceinline__ __bf16 f2bf(float f) {
  union { float f; unsigned u; } a; a.f = f;
  unsigned r = a.u + 0x7FFFu + ((a.u >> 16) & 1u);   // round-to-nearest-even
  union { unsigned short s; __bf16 b; } o; o.s = (unsigned short)(r >> 16);
  return o.b;
}

// ---------- pack fp32 weight (K x N row-major) into CDNA5 B-fragment layout ----------
// dest index: (k>>5)*32*N + n*32 + (k&31)  -> column n's 32 k contiguous; lane reads 8 at half*8 and 8 at 16+half*8
__global__ __launch_bounds__(256) void pack_w(const float* __restrict__ W,
                                              __bf16* __restrict__ P, int K, int N) {
  int t = blockIdx.x * blockDim.x + threadIdx.x;
  if (t >= K * N) return;
  int k = t / N, n = t - k * N;
  P[(size_t)(k >> 5) * 32 * N + (size_t)n * 32 + (k & 31)] = f2bf(W[t]);
}

// ---------- patchify: image (64^3) -> A (4096 x 64) bf16, einsum nctphqwr->nthwpqrc ----------
__global__ __launch_bounds__(256) void patch_gather(const float* __restrict__ img,
                                                    __bf16* __restrict__ A) {
  int t = blockIdx.x * blockDim.x + threadIdx.x;
  if (t >= NPATCH * PVOL) return;
  int n = t >> 6, p = t & 63;
  int tc = n >> 8, h = (n >> 4) & 15, w = n & 15;
  int pp = p >> 4, qq = (p >> 2) & 3, rr = p & 3;
  int vox = ((tc * 4 + pp) * IMGX + (h * 4 + qq)) * IMGX + (w * 4 + rr);
  A[t] = f2bf(img[vox]);
}

// ---------- bos / eos token embeddings ----------
__global__ __launch_bounds__(256) void embed_ends(const int* __restrict__ ids,
                                                  const float* __restrict__ emb,
                                                  float* __restrict__ x) {
  int t = threadIdx.x;
  x[t] = emb[(size_t)ids[0] * DMODEL + t];
  x[(size_t)(LSEQ - 1) * DMODEL + t] = emb[(size_t)ids[LSEQ - 1] * DMODEL + t];
}

// ---------- layernorm: one 256-thread block per row ----------
__global__ __launch_bounds__(256) void ln_row(const float* __restrict__ x,
                                              const float* __restrict__ s,
                                              const float* __restrict__ b,
                                              __bf16* outBf, float* outF) {
  __shared__ float red[DMODEL];
  int row = blockIdx.x, tid = threadIdx.x;
  float v = x[(size_t)row * DMODEL + tid];
  red[tid] = v; __syncthreads();
  for (int st = 128; st > 0; st >>= 1) { if (tid < st) red[tid] += red[tid + st]; __syncthreads(); }
  float mean = red[0] * (1.0f / DMODEL); __syncthreads();
  float c = v - mean;
  red[tid] = c * c; __syncthreads();
  for (int st = 128; st > 0; st >>= 1) { if (tid < st) red[tid] += red[tid + st]; __syncthreads(); }
  float var = red[0] * (1.0f / DMODEL);
  float y = c * rsqrtf(var + 1e-5f) * s[tid] + b[tid];
  if (outBf) outBf[(size_t)row * DMODEL + tid] = f2bf(y);
  if (outF)  outF [(size_t)row * DMODEL + tid] = y;
}

// ---------- bf16 WMMA GEMM: C = A(MxK) @ Bpk(KxN packed) (+bias)(+gelu)(+resid) ----------
// 8 waves/block tile 16 rows x 128 cols. A tile (16x32 bf16, 1 KB) is staged
// global -> LDS with double-buffered CDNA5 async-LDS b128 loads (ASYNCcnt):
// copy of tile it+1 overlaps WMMA of tile it; one barrier per iteration.
__global__ __launch_bounds__(256) void gemm_wmma_bf16(const __bf16* __restrict__ A,
                                                      const __bf16* __restrict__ Bpk,
                                                      const float* __restrict__ bias,
                                                      const float* __restrict__ resid,
                                                      float* __restrict__ outF,
                                                      __bf16* __restrict__ outBf,
                                                      int M, int K, int N, int doGelu) {
  __shared__ __bf16 ldsA[2][16 * 32];   // sole shared object -> group-segment offset 0
  int tid  = threadIdx.x;
  int wave = tid >> 5;
  int lane = tid & 31;
  int rt   = blockIdx.x;
  int ct   = blockIdx.y * 8 + wave;
  int half = lane >> 4;
  int l16  = lane & 15;

  // async-copy mapping: waves 0-1 (tid<64) each move 16 bytes per tile
  int cRow   = tid >> 2;               // 0..15 (for tid < 64)
  int cChunk = (tid & 3) * 8;          // element offset within 32-wide k tile
  int gRow = rt * 16 + (cRow & 15); if (gRow >= M) gRow = M - 1;   // clamp loads
  const __bf16* gsrc = A + (size_t)gRow * K + cChunk;
  unsigned ldsBase = (unsigned)((cRow & 15) * 32 + cChunk) * 2u;   // LDS byte offset

  int nT = K >> 5;
  if (tid < 64) {  // prime the pipeline: tile 0 -> buffer 0
    unsigned long long ga = (unsigned long long)(uintptr_t)gsrc;
    asm volatile("global_load_async_to_lds_b128 %0, %1, off"
                 :: "v"(ldsBase), "v"(ga) : "memory");
  }

  int col = ct * 16 + l16;
  v8f c = {};
  for (int it = 0; it < nT; ++it) {
    asm volatile("s_wait_asynccnt 0x0" ::: "memory");  // current tile's copy done (per wave)
    __syncthreads();                                   // visible to all; other buffer free
    if (it + 1 < nT && tid < 64) {                     // overlap: copy tile it+1
      unsigned long long ga = (unsigned long long)(uintptr_t)(gsrc + (it + 1) * 32);
      unsigned ld = ldsBase + (unsigned)(((it + 1) & 1) << 10);
      asm volatile("global_load_async_to_lds_b128 %0, %1, off"
                   :: "v"(ld), "v"(ga) : "memory");
    }
    union { v16bf v; v8bf h[2]; } af, bfv;
    // A fragment from LDS: elems 0-7 = K half*8+0..7 ; elems 8-15 = K 16+half*8+0..7
    const __bf16* ap = &ldsA[it & 1][l16 * 32 + half * 8];
    af.h[0] = *(const v8bf*)(ap);
    af.h[1] = *(const v8bf*)(ap + 16);
    // B fragment: col = ct*16 + (lane&15), elems 0-7 = K half*8+0..7 ; elems 8-15 = K 16+half*8+0..7
    const __bf16* bp = Bpk + (size_t)it * 32 * N + (size_t)col * 32 + half * 8;
    __builtin_prefetch(bp + (size_t)32 * N, 0, 1);     // next k-tile B line
    bfv.h[0] = *(const v8bf*)(bp);
    bfv.h[1] = *(const v8bf*)(bp + 16);
    c = __builtin_amdgcn_wmma_f32_16x16x32_bf16(false, af.v, false, bfv.v,
                                                (short)0, c, false, false);
  }
  // C/D layout: n = ct*16 + (lane&15), VGPR r -> m = rt*16 + half*8 + r
  int mbase = rt * 16 + half * 8;
  int n = ct * 16 + l16;
  float bn = bias ? bias[n] : 0.0f;
  for (int r = 0; r < 8; ++r) {
    int m = mbase + r;
    if (m >= M) continue;
    float v = c[r] + bn;
    if (doGelu) v = 0.5f * v * (1.0f + erff(v * 0.70710678118f));  // exact GELU
    if (resid) v += resid[(size_t)m * N + n];
    if (outF)  outF [(size_t)m * N + n] = v;
    if (outBf) outBf[(size_t)m * N + n] = f2bf(v);
  }
}

// ---------- sparse attention: one wave per (token, head), online softmax ----------
__global__ __launch_bounds__(256) void attn_sparse(const float* __restrict__ q,
                                                   const float* __restrict__ k,
                                                   const float* __restrict__ v,
                                                   const int* __restrict__ idx,
                                                   const float* __restrict__ geo,
                                                   const float* __restrict__ decay,
                                                   __bf16* __restrict__ aoBf, int Kc) {
  int gw = blockIdx.x * 8 + (threadIdx.x >> 5);
  if (gw >= LSEQ * NHEAD) return;
  int i    = gw >> 2;       // / NHEAD
  int head = gw & 3;
  int lane = threadIdx.x & 31;
  int d0   = head * DHEAD + lane;
  const float scale = 0.125f;          // 1/sqrt(64)
  float q0 = q[(size_t)i * DMODEL + d0]      * scale;
  float q1 = q[(size_t)i * DMODEL + d0 + 32] * scale;
  float dec = decay[head];
  float m = -__builtin_inff(), ssum = 0.0f, a0 = 0.0f, a1 = 0.0f;
  for (int j = 0; j < Kc; ++j) {
    int kidx = idx[(size_t)i * Kc + j];
    // slots 0,1 always valid (bos/eos); slot>=2 with kidx==0 is padding; causal: kidx<=i
    bool ok = ((j < 2) || (kidx != 0)) && (kidx <= i);
    if (!ok) continue;
    const float* kr = k + (size_t)kidx * DMODEL;
    float p = q0 * kr[d0] + q1 * kr[d0 + 32];
    for (int off = 16; off > 0; off >>= 1) p += __shfl_xor(p, off, 32);  // wave32 butterfly
    float logit = p + geo[(size_t)i * Kc + j] * dec;
    float nm   = fmaxf(m, logit);
    float corr = __expf(m - nm);
    float pe   = __expf(logit - nm);
    const float* vr = v + (size_t)kidx * DMODEL;
    ssum = ssum * corr + pe;
    a0   = a0   * corr + pe * vr[d0];
    a1   = a1   * corr + pe * vr[d0 + 32];
    m = nm;
  }
  float inv = 1.0f / ssum;
  aoBf[(size_t)i * DMODEL + d0]      = f2bf(a0 * inv);
  aoBf[(size_t)i * DMODEL + d0 + 32] = f2bf(a1 * inv);
}

// ---------- host ----------
extern "C" void kernel_launch(void* const* d_in, const int* in_sizes, int n_in,
                              void* d_out, int out_size, void* d_ws, size_t ws_size,
                              hipStream_t stream) {
  (void)n_in; (void)out_size; (void)ws_size;
  const int*   ids    = (const int*)  d_in[0];
  const float* image  = (const float*)d_in[1];
  const int*   idx    = (const int*)  d_in[2];
  /* d_in[3] = valid (bool) -- derived on device instead */
  const float* geo    = (const float*)d_in[4];
  const float* decay  = (const float*)d_in[5];
  const float* embed  = (const float*)d_in[6];
  const float* patchw = (const float*)d_in[7];
  const float* patchb = (const float*)d_in[8];
  const float* wq  = (const float*)d_in[9];
  const float* wk  = (const float*)d_in[10];
  const float* wv  = (const float*)d_in[11];
  const float* wo  = (const float*)d_in[12];
  const float* bo  = (const float*)d_in[13];
  const float* ln1s = (const float*)d_in[14];
  const float* ln1b = (const float*)d_in[15];
  const float* ln2s = (const float*)d_in[16];
  const float* ln2b = (const float*)d_in[17];
  const float* w1  = (const float*)d_in[18];
  const float* b1  = (const float*)d_in[19];
  const float* w2  = (const float*)d_in[20];
  const float* b2  = (const float*)d_in[21];
  const float* norms = (const float*)d_in[22];
  const float* normb = (const float*)d_in[23];

  const int Kc = in_sizes[2] / LSEQ;   // neighbors per row

  char* ws = (char*)d_ws;
  size_t off = 0;
  auto alloc = [&](size_t bytes) -> void* {
    void* p = (void*)(ws + off);
    off = (off + bytes + 255) & ~((size_t)255);
    return p;
  };

  __bf16* pw_pk = (__bf16*)alloc((size_t)PVOL * DMODEL * 2);
  __bf16 *wq_pk[NLAY], *wk_pk[NLAY], *wv_pk[NLAY], *wo_pk[NLAY], *w1_pk[NLAY], *w2_pk[NLAY];
  for (int l = 0; l < NLAY; ++l) {
    wq_pk[l] = (__bf16*)alloc((size_t)DMODEL * DMODEL * 2);
    wk_pk[l] = (__bf16*)alloc((size_t)DMODEL * DMODEL * 2);
    wv_pk[l] = (__bf16*)alloc((size_t)DMODEL * DMODEL * 2);
    wo_pk[l] = (__bf16*)alloc((size_t)DMODEL * DMODEL * 2);
    w1_pk[l] = (__bf16*)alloc((size_t)DMODEL * DFFN * 2);
    w2_pk[l] = (__bf16*)alloc((size_t)DFFN * DMODEL * 2);
  }
  float*  x      = (float*) alloc((size_t)LSEQ * DMODEL * 4);
  __bf16* hBf    = (__bf16*)alloc((size_t)LSEQ * DMODEL * 2);
  float*  qF     = (float*) alloc((size_t)LSEQ * DMODEL * 4);
  float*  kF     = (float*) alloc((size_t)LSEQ * DMODEL * 4);
  float*  vF     = (float*) alloc((size_t)LSEQ * DMODEL * 4);
  __bf16* aoBf   = (__bf16*)alloc((size_t)LSEQ * DMODEL * 2);
  __bf16* gBf    = (__bf16*)alloc((size_t)LSEQ * DFFN   * 2);
  __bf16* aPatch = (__bf16*)alloc((size_t)NPATCH * PVOL * 2);

  auto packW = [&](const float* W, __bf16* P, int K, int N) {
    int tot = K * N;
    pack_w<<<(tot + 255) / 256, 256, 0, stream>>>(W, P, K, N);
  };
  packW(patchw, pw_pk, PVOL, DMODEL);
  for (int l = 0; l < NLAY; ++l) {
    packW(wq + (size_t)l * DMODEL * DMODEL, wq_pk[l], DMODEL, DMODEL);
    packW(wk + (size_t)l * DMODEL * DMODEL, wk_pk[l], DMODEL, DMODEL);
    packW(wv + (size_t)l * DMODEL * DMODEL, wv_pk[l], DMODEL, DMODEL);
    packW(wo + (size_t)l * DMODEL * DMODEL, wo_pk[l], DMODEL, DMODEL);
    packW(w1 + (size_t)l * DMODEL * DFFN,   w1_pk[l], DMODEL, DFFN);
    packW(w2 + (size_t)l * DFFN * DMODEL,   w2_pk[l], DFFN, DMODEL);
  }

  patch_gather<<<(NPATCH * PVOL + 255) / 256, 256, 0, stream>>>(image, aPatch);

  auto gemm = [&](const __bf16* A, const __bf16* B, const float* bias, const float* resid,
                  float* oF, __bf16* oB, int M, int K, int N, int gelu) {
    dim3 grid((M + 15) / 16, N / 128);
    gemm_wmma_bf16<<<grid, 256, 0, stream>>>(A, B, bias, resid, oF, oB, M, K, N, gelu);
  };

  // patch embed -> x rows 1..4096 ; bos/eos -> rows 0, L-1
  gemm(aPatch, pw_pk, patchb, nullptr, x + DMODEL, nullptr, NPATCH, PVOL, DMODEL, 0);
  embed_ends<<<1, 256, 0, stream>>>(ids, embed, x);

  for (int l = 0; l < NLAY; ++l) {
    ln_row<<<LSEQ, 256, 0, stream>>>(x, ln1s + l * DMODEL, ln1b + l * DMODEL, hBf, nullptr);
    gemm(hBf, wq_pk[l], nullptr, nullptr, qF, nullptr, LSEQ, DMODEL, DMODEL, 0);
    gemm(hBf, wk_pk[l], nullptr, nullptr, kF, nullptr, LSEQ, DMODEL, DMODEL, 0);
    gemm(hBf, wv_pk[l], nullptr, nullptr, vF, nullptr, LSEQ, DMODEL, DMODEL, 0);
    attn_sparse<<<(LSEQ * NHEAD + 7) / 8, 256, 0, stream>>>(qF, kF, vF, idx, geo, decay, aoBf, Kc);
    gemm(aoBf, wo_pk[l], bo + l * DMODEL, x, x, nullptr, LSEQ, DMODEL, DMODEL, 0);
    ln_row<<<LSEQ, 256, 0, stream>>>(x, ln2s + l * DMODEL, ln2b + l * DMODEL, hBf, nullptr);
    gemm(hBf, w1_pk[l], b1 + l * DFFN, nullptr, nullptr, gBf, LSEQ, DMODEL, DFFN, 1);
    gemm(gBf, w2_pk[l], b2 + l * DMODEL, x, x, nullptr, LSEQ, DFFN, DMODEL, 0);
  }
  ln_row<<<LSEQ, 256, 0, stream>>>(x, norms, normb, nullptr, (float*)d_out);
}
